// VerticalStripAttention_42649025249311
// MI455X (gfx1250) — matrix operand order repaired and checked by reading, not compiled, not run on hardware
//
#include <hip/hip_runtime.h>

typedef __attribute__((ext_vector_type(16))) _Float16 v16h;
typedef __attribute__((ext_vector_type(8)))  float    v8f;
typedef __attribute__((ext_vector_type(4)))  int      v4i;

union AF { v16h v; uint4 u[2]; _Float16 h[16]; };
union CF { v8f  v; float f[8]; };

#define DEVFN __device__ __forceinline__

#if defined(__HIP_DEVICE_COMPILE__) && \
    __has_builtin(__builtin_amdgcn_global_load_async_to_lds_b128) && \
    __has_builtin(__builtin_amdgcn_s_wait_asynccnt)
#define HAVE_ASYNC_LDS 1
#else
#define HAVE_ASYNC_LDS 0
#endif

// ---- problem constants ----
constexpr int   NTOK   = 256;
constexpr int   DIMC   = 192;
constexpr int   HEADS  = 6;
constexpr int   BATCH  = 256;           // B_
constexpr int   NS     = 64;
constexpr int   TOKENS = BATCH * NTOK;  // 65536
constexpr float SCALE  = 0.17677669529663687f; // 32^-0.5

// ---- workspace layout (bytes) ----
constexpr size_t OFF_QKVW = 0;                                   // 576*192 f16
constexpr size_t OFF_PROJW = OFF_QKVW + (size_t)576 * 192 * 2;   // 192*192 f16
constexpr size_t OFF_BIAS6 = OFF_PROJW + (size_t)192 * 192 * 2;  // 6*256*256 f32
constexpr size_t OFF_Q  = OFF_BIAS6 + (size_t)6 * 65536 * 4;     // [b*6+h][256][32] f16
constexpr size_t OFF_K  = OFF_Q + (size_t)TOKENS * DIMC * 2;     // [b*6+h][256][32] f16
constexpr size_t OFF_VT = OFF_K + (size_t)TOKENS * DIMC * 2;     // [b*6+h][32][256] f16
constexpr size_t OFF_AO = OFF_VT + (size_t)TOKENS * DIMC * 2;    // [token][192] f16

DEVFN v8f wmma_f16(const AF& a, const AF& b, v8f c) {
  return __builtin_amdgcn_wmma_f32_16x16x32_f16(false, a.v, false, b.v,
                                                (short)0, c, false, false);
}

#if HAVE_ASYNC_LDS
DEVFN void async_cp16(void* lds, const void* glob) {
  __builtin_amdgcn_global_load_async_to_lds_b128(
      (__attribute__((address_space(1))) v4i*)(glob),
      (__attribute__((address_space(3))) v4i*)(lds), 0, 0);
}
#endif

DEVFN float redmax16(float v) {
  v = fmaxf(v, __shfl_xor(v, 1, 32));
  v = fmaxf(v, __shfl_xor(v, 2, 32));
  v = fmaxf(v, __shfl_xor(v, 4, 32));
  v = fmaxf(v, __shfl_xor(v, 8, 32));
  return v;
}
DEVFN float redsum16(float v) {
  v += __shfl_xor(v, 1, 32);
  v += __shfl_xor(v, 2, 32);
  v += __shfl_xor(v, 4, 32);
  v += __shfl_xor(v, 8, 32);
  return v;
}

// ---------------------------------------------------------------------------
// Kernel 0: weight f32->f16 casts + relative-position-bias pre-gather
// ---------------------------------------------------------------------------
__global__ __launch_bounds__(256) void prep_kernel(
    const float* __restrict__ qkv_w, const float* __restrict__ proj_w,
    const float* __restrict__ rpb_table, const int* __restrict__ rel_index,
    _Float16* __restrict__ qkvw_h, _Float16* __restrict__ projw_h,
    float* __restrict__ bias6) {
  int idx = blockIdx.x * 256 + threadIdx.x;
  if (idx < 576 * 192) qkvw_h[idx] = (_Float16)qkv_w[idx];
  int r2 = idx - 576 * 192;
  if (r2 >= 0 && r2 < 192 * 192) projw_h[r2] = (_Float16)proj_w[r2];
  int r3 = idx - (576 * 192 + 192 * 192);
  if (r3 >= 0 && r3 < 6 * 65536) {
    int h = r3 >> 16, ij = r3 & 65535;
    bias6[r3] = rpb_table[rel_index[ij] * HEADS + h];
  }
}

// ---------------------------------------------------------------------------
// Kernel 1: QKV GEMM  (x[65536,192] @ qkv_w^T[192,576] + qkv_b)
// 8 waves/block, wave = one 16-token M tile. Two N tiles per iteration
// (independent WMMA chains). Weight pair-tile (32x192 f16) staged in LDS.
// ---------------------------------------------------------------------------
__global__ __launch_bounds__(256) void qkv_kernel(
    const float* __restrict__ x, const float* __restrict__ qkv_b,
    const _Float16* __restrict__ qkvw_h,
    _Float16* __restrict__ Qb, _Float16* __restrict__ Kb,
    _Float16* __restrict__ Vt) {
  __shared__ __align__(16) _Float16 wlds[32 * 192];  // 12 KB
  int wave = threadIdx.x >> 5, lane = threadIdx.x & 31;
  int half = lane >> 4, ln = lane & 15;
  int token_base = (blockIdx.x * 8 + wave) * 16;

  // A fragments: 6 K-chunks of 32, f32 -> f16 on load
  AF a[6];
  const float* xr = x + (size_t)(token_base + ln) * DIMC;
#pragma unroll
  for (int kc = 0; kc < 6; kc++) {
    const float* p0 = xr + kc * 32 + 16 * half;
    const float* p1 = xr + kc * 32 + 16 * half + 8;
#pragma unroll
    for (int j = 0; j < 8; j++) {
      a[kc].h[j]     = (_Float16)p0[j];
      a[kc].h[8 + j] = (_Float16)p1[j];
    }
  }

  for (int ntp = 0; ntp < 18; ntp++) {  // pairs of 16-col N tiles
    __syncthreads();
    {
      const uint4* src = (const uint4*)(qkvw_h + (size_t)ntp * 32 * 192);
      uint4* dst = (uint4*)wlds;
#pragma unroll
      for (int i = 0; i < 2; i++) {
        int idx = threadIdx.x + 256 * i;
        if (idx < 384) {
#if HAVE_ASYNC_LDS
          async_cp16(dst + idx, src + idx);
#else
          dst[idx] = src[idx];
#endif
        }
      }
#if HAVE_ASYNC_LDS
      __builtin_amdgcn_s_wait_asynccnt(0);
#endif
    }
    __syncthreads();

    int n0 = ntp * 32 + ln;
    CF c0, c1;
    float b0 = qkv_b[n0], b1 = qkv_b[n0 + 16];
#pragma unroll
    for (int r = 0; r < 8; r++) { c0.f[r] = b0; c1.f[r] = b1; }

    AF bf0[6], bf1[6];
#pragma unroll
    for (int kc = 0; kc < 6; kc++) {
      const _Float16* w0 = wlds + ln * 192 + kc * 32 + 16 * half;
      const _Float16* w1 = w0 + 16 * 192;
      bf0[kc].u[0] = *(const uint4*)w0;
      bf0[kc].u[1] = *(const uint4*)(w0 + 8);
      bf1[kc].u[0] = *(const uint4*)w1;
      bf1[kc].u[1] = *(const uint4*)(w1 + 8);
    }
#pragma unroll
    for (int kc = 0; kc < 6; kc++) {
      c0.v = wmma_f16(a[kc], bf0[kc], c0.v);
      c1.v = wmma_f16(a[kc], bf1[kc], c1.v);
    }

    // Uniform routing: which of q/k/v, head = ntp % 6, hd = ln (+16 for c1)
    int which = ntp / 6, hh = ntp % 6;
    if (which == 0) {
#pragma unroll
      for (int r = 0; r < 8; r++) {
        int tok = token_base + r + 8 * half;
        size_t base = ((size_t)((tok >> 8) * HEADS + hh) * 256 + (tok & 255)) * 32;
        Qb[base + ln]      = (_Float16)(c0.f[r] * SCALE);
        Qb[base + 16 + ln] = (_Float16)(c1.f[r] * SCALE);
      }
    } else if (which == 1) {
#pragma unroll
      for (int r = 0; r < 8; r++) {
        int tok = token_base + r + 8 * half;
        size_t base = ((size_t)((tok >> 8) * HEADS + hh) * 256 + (tok & 255)) * 32;
        Kb[base + ln]      = (_Float16)c0.f[r];
        Kb[base + 16 + ln] = (_Float16)c1.f[r];
      }
    } else {
#pragma unroll
      for (int r = 0; r < 8; r++) {
        int tok = token_base + r + 8 * half;
        int tt = tok & 255;
        size_t bhh = (size_t)((tok >> 8) * HEADS + hh);
        Vt[(bhh * 32 + ln) * 256 + tt]      = (_Float16)c0.f[r];
        Vt[(bhh * 32 + 16 + ln) * 256 + tt] = (_Float16)c1.f[r];
      }
    }
  }
}

// ---------------------------------------------------------------------------
// Kernel 2: attention. block = (b,h, half of query tiles), 8 waves.
// K + V^T staged in LDS (async); wave owns a 16-query row block; full
// 256-key logits in registers; softmax; PV via per-wave LDS prob staging.
// Dynamic LDS: K 16KB | Vt 16KB | probs 8 x 8KB = 96KB.
// ---------------------------------------------------------------------------
__global__ __launch_bounds__(256) void attn_kernel(
    const float* __restrict__ mask, const float* __restrict__ bias6,
    const _Float16* __restrict__ Qb, const _Float16* __restrict__ Kb,
    const _Float16* __restrict__ Vt, _Float16* __restrict__ ao) {
  extern __shared__ __align__(16) _Float16 smem[];
  _Float16* klds = smem;                 // [256][32]
  _Float16* vlds = smem + 8192;          // [32][256]
  int wave = threadIdx.x >> 5, lane = threadIdx.x & 31;
  int half = lane >> 4, ln = lane & 15;
  _Float16* pw = smem + 16384 + wave * 4096;  // [16][256] per wave

  int bh = blockIdx.x >> 1, qg = blockIdx.x & 1;
  int b = bh / HEADS, h = bh % HEADS, s = b & (NS - 1);
  int qbase = (qg * 8 + wave) * 16;

  {
    const uint4* kg = (const uint4*)(Kb + (size_t)bh * 8192);
    const uint4* vg = (const uint4*)(Vt + (size_t)bh * 8192);
    uint4* kl = (uint4*)klds;
    uint4* vl = (uint4*)vlds;
#pragma unroll
    for (int i = 0; i < 4; i++) {
      int idx = threadIdx.x + 256 * i;
#if HAVE_ASYNC_LDS
      async_cp16(kl + idx, kg + idx);
      async_cp16(vl + idx, vg + idx);
#else
      kl[idx] = kg[idx];
      vl[idx] = vg[idx];
#endif
    }
  }

  // Q fragment while staging is in flight
  AF a;
  const _Float16* qrow = Qb + (size_t)bh * 8192 + (size_t)(qbase + ln) * 32;
  a.u[0] = *(const uint4*)(qrow + 16 * half);
  a.u[1] = *(const uint4*)(qrow + 16 * half + 8);

#if HAVE_ASYNC_LDS
  __builtin_amdgcn_s_wait_asynccnt(0);
#endif
  __syncthreads();

  // logits: 16 tiles of 16x16 f32, prefetch-next pipeline on B fragments
  CF st[16];
  const float* brow = bias6 + (size_t)h * 65536;
  const float* mrow = mask + (size_t)s * 65536;
  AF bf;
  {
    const _Float16* kp = klds + ln * 32 + 16 * half;
    bf.u[0] = *(const uint4*)kp;
    bf.u[1] = *(const uint4*)(kp + 8);
  }
#pragma unroll
  for (int kt = 0; kt < 16; kt++) {
    AF cur = bf;
    if (kt < 15) {
      const _Float16* kp = klds + ((kt + 1) * 16 + ln) * 32 + 16 * half;
      bf.u[0] = *(const uint4*)kp;
      bf.u[1] = *(const uint4*)(kp + 8);
    }
    v8f cz = {};
    CF c;
    c.v = wmma_f16(a, cur, cz);
    int j = kt * 16 + ln;
#pragma unroll
    for (int r = 0; r < 8; r++) {
      int i = qbase + r + 8 * half;
      c.f[r] += brow[i * 256 + j] + mrow[i * 256 + j];
    }
    st[kt] = c;
  }

  // row-wise softmax (rows of a C/D tile live in 16-lane halves)
  float mx[8], sm[8], inv[8];
#pragma unroll
  for (int r = 0; r < 8; r++) mx[r] = st[0].f[r];
#pragma unroll
  for (int kt = 1; kt < 16; kt++)
#pragma unroll
    for (int r = 0; r < 8; r++) mx[r] = fmaxf(mx[r], st[kt].f[r]);
#pragma unroll
  for (int r = 0; r < 8; r++) { mx[r] = redmax16(mx[r]); sm[r] = 0.0f; }

#pragma unroll
  for (int kt = 0; kt < 16; kt++) {
#pragma unroll
    for (int r = 0; r < 8; r++) {
      float p = __expf(st[kt].f[r] - mx[r]);
      sm[r] += p;
      pw[(r + 8 * half) * 256 + kt * 16 + ln] = (_Float16)p;
    }
  }
#pragma unroll
  for (int r = 0; r < 8; r++) inv[r] = 1.0f / redsum16(sm[r]);

  // out = P @ V : two independent N-tile chains share the A fragment
  CF c0, c1;
  {
    v8f z = {};
    c0.v = z;
    c1.v = z;
  }
#pragma unroll
  for (int kc = 0; kc < 8; kc++) {
    AF pa, vb0, vb1;
    const _Float16* pp = pw + ln * 256 + kc * 32;
    pa.u[0] = *(const uint4*)(pp + 16 * half);
    pa.u[1] = *(const uint4*)(pp + 16 * half + 8);
    const _Float16* vp0 = vlds + ln * 256 + kc * 32 + 16 * half;
    const _Float16* vp1 = vp0 + 16 * 256;
    vb0.u[0] = *(const uint4*)vp0;
    vb0.u[1] = *(const uint4*)(vp0 + 8);
    vb1.u[0] = *(const uint4*)vp1;
    vb1.u[1] = *(const uint4*)(vp1 + 8);
    c0.v = wmma_f16(pa, vb0, c0.v);
    c1.v = wmma_f16(pa, vb1, c1.v);
  }
#pragma unroll
  for (int r = 0; r < 8; r++) {
    int tok = qbase + r + 8 * half;
    size_t base = ((size_t)b * 256 + tok) * 192 + h * 32;
    ao[base + ln]      = (_Float16)(c0.f[r] * inv[r]);
    ao[base + 16 + ln] = (_Float16)(c1.f[r] * inv[r]);
  }
}

// ---------------------------------------------------------------------------
// Kernel 3: output projection  (ao[65536,192] @ proj_w^T[192,192] + proj_b)
// Same scheme as qkv: two N tiles per iteration.
// ---------------------------------------------------------------------------
__global__ __launch_bounds__(256) void proj_kernel(
    const _Float16* __restrict__ ao, const float* __restrict__ proj_b,
    const _Float16* __restrict__ projw_h, float* __restrict__ out) {
  __shared__ __align__(16) _Float16 wlds[32 * 192];
  int wave = threadIdx.x >> 5, lane = threadIdx.x & 31;
  int half = lane >> 4, ln = lane & 15;
  int token_base = (blockIdx.x * 8 + wave) * 16;

  AF a[6];
  const _Float16* ar = ao + (size_t)(token_base + ln) * 192;
#pragma unroll
  for (int kc = 0; kc < 6; kc++) {
    a[kc].u[0] = *(const uint4*)(ar + kc * 32 + 16 * half);
    a[kc].u[1] = *(const uint4*)(ar + kc * 32 + 16 * half + 8);
  }

  for (int ntp = 0; ntp < 6; ntp++) {
    __syncthreads();
    {
      const uint4* src = (const uint4*)(projw_h + (size_t)ntp * 32 * 192);
      uint4* dst = (uint4*)wlds;
#pragma unroll
      for (int i = 0; i < 2; i++) {
        int idx = threadIdx.x + 256 * i;
        if (idx < 384) {
#if HAVE_ASYNC_LDS
          async_cp16(dst + idx, src + idx);
#else
          dst[idx] = src[idx];
#endif
        }
      }
#if HAVE_ASYNC_LDS
      __builtin_amdgcn_s_wait_asynccnt(0);
#endif
    }
    __syncthreads();

    int n0 = ntp * 32 + ln;
    CF c0, c1;
    float b0 = proj_b[n0], b1 = proj_b[n0 + 16];
#pragma unroll
    for (int r = 0; r < 8; r++) { c0.f[r] = b0; c1.f[r] = b1; }

    AF bf0[6], bf1[6];
#pragma unroll
    for (int kc = 0; kc < 6; kc++) {
      const _Float16* w0 = wlds + ln * 192 + kc * 32 + 16 * half;
      const _Float16* w1 = w0 + 16 * 192;
      bf0[kc].u[0] = *(const uint4*)w0;
      bf0[kc].u[1] = *(const uint4*)(w0 + 8);
      bf1[kc].u[0] = *(const uint4*)w1;
      bf1[kc].u[1] = *(const uint4*)(w1 + 8);
    }
#pragma unroll
    for (int kc = 0; kc < 6; kc++) {
      c0.v = wmma_f16(a[kc], bf0[kc], c0.v);
      c1.v = wmma_f16(a[kc], bf1[kc], c1.v);
    }

#pragma unroll
    for (int r = 0; r < 8; r++) {
      size_t row = (size_t)(token_base + r + 8 * half) * 192;
      out[row + n0]      = c0.f[r];
      out[row + n0 + 16] = c1.f[r];
    }
  }
}

// ---------------------------------------------------------------------------
extern "C" void kernel_launch(void* const* d_in, const int* in_sizes, int n_in,
                              void* d_out, int out_size, void* d_ws,
                              size_t ws_size, hipStream_t stream) {
  (void)in_sizes; (void)n_in; (void)out_size; (void)ws_size;
  const float* x      = (const float*)d_in[0];
  const float* mask   = (const float*)d_in[1];
  const float* qkv_w  = (const float*)d_in[2];
  const float* qkv_b  = (const float*)d_in[3];
  const float* proj_w = (const float*)d_in[4];
  const float* proj_b = (const float*)d_in[5];
  const float* rpb    = (const float*)d_in[6];
  const int*   rel    = (const int*)d_in[7];

  char* ws = (char*)d_ws;
  _Float16* qkvw_h = (_Float16*)(ws + OFF_QKVW);
  _Float16* projw_h = (_Float16*)(ws + OFF_PROJW);
  float*    bias6  = (float*)(ws + OFF_BIAS6);
  _Float16* Qb = (_Float16*)(ws + OFF_Q);
  _Float16* Kb = (_Float16*)(ws + OFF_K);
  _Float16* Vt = (_Float16*)(ws + OFF_VT);
  _Float16* ao = (_Float16*)(ws + OFF_AO);

  prep_kernel<<<2112, 256, 0, stream>>>(qkv_w, proj_w, rpb, rel,
                                        qkvw_h, projw_h, bias6);
  qkv_kernel<<<TOKENS / 16 / 8, 256, 0, stream>>>(x, qkv_b, qkvw_h, Qb, Kb, Vt);
  attn_kernel<<<BATCH * HEADS * 2, 256, 96 * 1024, stream>>>(mask, bias6, Qb,
                                                             Kb, Vt, ao);
  proj_kernel<<<TOKENS / 16 / 8, 256, 0, stream>>>(ao, proj_b, projw_h,
                                                   (float*)d_out);
}
